// MultiHeadAttention_6365141533163
// MI455X (gfx1250) — matrix operand order repaired and checked by reading, not compiled, not run on hardware
//
#include <hip/hip_runtime.h>

// ---------------------------------------------------------------------------
// Multi-head spatial attention for MI455X (gfx1250, wave32, WMMA).
// B=8 T=64 N=170 C=128 H=8 D=16.  All GEMMs use v_wmma_f32_16x16x32_bf16.
// Workspace layout (bytes):
//   [0,131072)        : Wv|Wk|Wq|Wo as bf16 (4 x 128x128)
//   +22282240 each    : vproj, kproj, qproj  (bf16, [bt][h][170][16])
//   last 22282240     : attention output     (bf16, [row][128])
// Total ~89.3 MB.
// ---------------------------------------------------------------------------

typedef __attribute__((ext_vector_type(16))) __bf16 v16bf;
typedef __attribute__((ext_vector_type(8)))  float  v8f;

union Frag {
  v16bf v;
  unsigned short u[16];
};

__device__ __forceinline__ unsigned short f2bf(float f) {
  unsigned u = __float_as_uint(f);
  unsigned r = u + 0x7FFFu + ((u >> 16) & 1u);   // round-to-nearest-even
  return (unsigned short)(r >> 16);
}

__device__ __forceinline__ v8f wmma_bf16(v16bf a, v16bf b, v8f c) {
  return __builtin_amdgcn_wmma_f32_16x16x32_bf16(
      /*neg_a=*/false, a, /*neg_b=*/false, b,
      /*c_mod=*/(short)0, c, /*reuse_a=*/false, /*reuse_b=*/false);
}

#define NROWS   87040      // B*T*N
#define NNODE   170
#define NPAD    176
#define NTILES  11
#define CDIM    128
#define HEADS   8
#define HDIM    16

// ---------------------------------------------------------------------------
// Kernel 0: convert weight matrices f32 -> bf16
// ---------------------------------------------------------------------------
__global__ __launch_bounds__(256) void convert_weights(
    const float* __restrict__ Wv, const float* __restrict__ Wk,
    const float* __restrict__ Wq, const float* __restrict__ Wo,
    unsigned short* __restrict__ wbf) {
  int idx = blockIdx.x * 256 + threadIdx.x;    // 65536 total
  int which = idx >> 14;
  int off = idx & 16383;
  float f;
  if (which == 0)      f = Wv[off];
  else if (which == 1) f = Wk[off];
  else if (which == 2) f = Wq[off];
  else                 f = Wo[off];
  wbf[idx] = f2bf(f);
}

// ---------------------------------------------------------------------------
// Kernel 1: fused Q/K/V projections.  One block = 16 input rows, 8 waves.
// Wave w computes head-tile h=w for each of the 3 projections (K=128 -> 4 WMMA).
// ---------------------------------------------------------------------------
__global__ __launch_bounds__(256) void qkv_proj(
    const float* __restrict__ vals, const float* __restrict__ keys,
    const float* __restrict__ qry,
    const unsigned short* __restrict__ wbf,   // Wv | Wk | Wq bf16
    const float* __restrict__ bv, const float* __restrict__ bk,
    const float* __restrict__ bq,
    unsigned short* __restrict__ vb, unsigned short* __restrict__ kb,
    unsigned short* __restrict__ qb) {
  __shared__ unsigned short tin[3][16][CDIM];

  const int tid = threadIdx.x;
  const int rowbase = blockIdx.x * 16;

  for (int i = tid; i < 3 * 16 * CDIM; i += 256) {
    int tt = i >> 11;
    int rem = i & 2047;
    int row = rem >> 7, col = rem & 127;
    const float* src = (tt == 0) ? vals : (tt == 1) ? keys : qry;
    tin[tt][row][col] = f2bf(src[(size_t)(rowbase + row) * CDIM + col]);
  }
  __syncthreads();

  const int lane = tid & 31;
  const int wave = tid >> 5;          // == head tile h
  const int half = lane >> 4;
  const int l15  = lane & 15;
  const int kbaseA = half * 8;        // unused; A and B both use kb2
  const int kb2    = half * 16;       // B-matrix K base per half-wave
  const int h = wave;

#pragma unroll
  for (int p = 0; p < 3; ++p) {       // 0=V, 1=K, 2=Q
    const unsigned short* W = wbf + p * 16384;
    const float* bias_p = (p == 0) ? bv : (p == 1) ? bk : bq;
    unsigned short* dst = (p == 0) ? vb : (p == 1) ? kb : qb;

    v8f acc = {};
#pragma unroll
    for (int ks = 0; ks < 4; ++ks) {  // K = 128 in 4 x 32 steps
      Frag a, b;
#pragma unroll
      for (int i = 0; i < 16; ++i) {
        int koff = kb2 + i;
        a.u[i] = tin[p][l15][ks * 32 + koff];
      }
      int ncol = h * HDIM + l15;      // out[m][n] = sum_k x[m][k] * W[n][k]
      const unsigned short* wrow = W + ncol * CDIM + ks * 32 + kb2;
#pragma unroll
      for (int i = 0; i < 16; ++i) b.u[i] = wrow[i];
      acc = wmma_bf16(a.v, b.v, acc);
    }

    float bias = bias_p[h * HDIM + l15];
#pragma unroll
    for (int i = 0; i < 8; ++i) {     // C/D: VGPR i -> row i + 8*half
      int m = i + half * 8;
      int r = rowbase + m;
      int bt = r / NNODE;
      int node = r - bt * NNODE;
      size_t off = ((size_t)(bt * HEADS + h) * NNODE + node) * HDIM + l15;
      dst[off] = f2bf(acc[i] + bias);
    }
  }
}

// ---------------------------------------------------------------------------
// Kernel 2: attention per (b,t,h).  Block = 4 waves; wave handles q-tiles
// qi = wave, wave+4, wave+8.  Q/K/V staged in LDS (bf16, N padded to 176).
// ---------------------------------------------------------------------------
__global__ __launch_bounds__(128) void attention(
    const unsigned short* __restrict__ qb, const unsigned short* __restrict__ kbuf,
    const unsigned short* __restrict__ vbuf, unsigned short* __restrict__ ao) {
  __shared__ unsigned short Qt[NPAD][HDIM];
  __shared__ unsigned short Kt[NPAD][HDIM];
  __shared__ unsigned short Vt[NPAD][HDIM];
  __shared__ unsigned short Sw[4][16][NPAD];   // per-wave prob staging

  const int tid = threadIdx.x;
  const int bth = blockIdx.x;                  // (b*T+t)*H + h
  const size_t base = (size_t)bth * NNODE * HDIM;
  const int bt = bth >> 3;
  const int h  = bth & 7;

  for (int i = tid; i < 3 * NPAD * HDIM; i += 128) {
    int tt = i / (NPAD * HDIM);
    int rem = i - tt * (NPAD * HDIM);
    int node = rem >> 4, d = rem & 15;
    unsigned short val = 0;
    if (node < NNODE) {
      const unsigned short* src = (tt == 0) ? qb : (tt == 1) ? kbuf : vbuf;
      val = src[base + node * HDIM + d];
    }
    if (tt == 0)      Qt[node][d] = val;
    else if (tt == 1) Kt[node][d] = val;
    else              Vt[node][d] = val;
  }
  __syncthreads();

  const int lane = tid & 31;
  const int wave = tid >> 5;
  const int half = lane >> 4;
  const int l15  = lane & 15;
  const int kbaseA = half * 8;
  const int kb2    = half * 16;
  const float scale = 0.088388347648318447f;   // 1/sqrt(128)

  for (int qi = wave; qi < NTILES; qi += 4) {
    // ---- Q fragment: 16x16 zero-padded to 16x32 (K = head dim) ----
    Frag aq;
#pragma unroll
    for (int i = 0; i < 16; ++i) {
      int koff = kb2 + i;
      aq.u[i] = (koff < HDIM) ? Qt[qi * 16 + l15][koff] : (unsigned short)0;
    }

    // ---- S = Q K^T : one WMMA per k-tile ----
    v8f accS[NTILES];
#pragma unroll
    for (int kt = 0; kt < NTILES; ++kt) {
      Frag bK;
      int knode = kt * 16 + l15;               // B col = lane -> key node
#pragma unroll
      for (int i = 0; i < 16; ++i) {
        int kin = kb2 + i;                     // inner dim = d
        bK.u[i] = (kin < HDIM) ? Kt[knode][kin] : (unsigned short)0;
      }
      v8f z = {};
      accS[kt] = wmma_bf16(aq.v, bK.v, z);
    }

    // ---- scale, mask, row softmax (row = VGPR p + 8*half; cols across lanes)
    float rsum[8];
#pragma unroll
    for (int p = 0; p < 8; ++p) {
      float mx = -3.0e38f;
#pragma unroll
      for (int kt = 0; kt < NTILES; ++kt) {
        float s = accS[kt][p] * scale;
        int col = kt * 16 + l15;
        if (col >= NNODE) s = -1.0e30f;        // mask padded key nodes
        accS[kt][p] = s;
        mx = fmaxf(mx, s);
      }
#pragma unroll
      for (int m = 1; m <= 8; m <<= 1) mx = fmaxf(mx, __shfl_xor(mx, m, 32));
      float sum = 0.f;
#pragma unroll
      for (int kt = 0; kt < NTILES; ++kt) {
        float e = __expf(accS[kt][p] - mx);
        accS[kt][p] = e;
        sum += e;
      }
#pragma unroll
      for (int m = 1; m <= 8; m <<= 1) sum += __shfl_xor(sum, m, 32);
      rsum[p] = sum;
    }

    // ---- stage normalized probs to LDS (C/D layout -> A layout transpose)
#pragma unroll
    for (int p = 0; p < 8; ++p) {
      float rinv = 1.0f / rsum[p];
      int m = p + half * 8;
#pragma unroll
      for (int kt = 0; kt < NTILES; ++kt) {
        Sw[wave][m][kt * 16 + l15] = f2bf(accS[kt][p] * rinv);
      }
    }

    // ---- O = P V : inner dim 176 in 6 x 32 chunks ----
    v8f accO = {};
#pragma unroll
    for (int kc = 0; kc < 6; ++kc) {
      Frag aP, bV;
#pragma unroll
      for (int i = 0; i < 16; ++i) {
        int koff = kb2 + i;
        int kg = kc * 32 + koff;
        aP.u[i] = (kg < NPAD) ? Sw[wave][l15][kg] : (unsigned short)0;
      }
#pragma unroll
      for (int i = 0; i < 16; ++i) {
        int kg = kc * 32 + kb2 + i;
        bV.u[i] = (kg < NPAD) ? Vt[kg][l15] : (unsigned short)0;
      }
      accO = wmma_bf16(aP.v, bV.v, accO);
    }

    // ---- store head output rows (bf16, [row][128]) ----
#pragma unroll
    for (int p = 0; p < 8; ++p) {
      int m = p + half * 8;
      int qnode = qi * 16 + m;
      if (qnode < NNODE) {
        size_t rg = (size_t)bt * NNODE + qnode;
        ao[rg * CDIM + h * HDIM + l15] = f2bf(accO[p]);
      }
    }
  }
}

// ---------------------------------------------------------------------------
// Kernel 3: output projection -> f32 d_out
// ---------------------------------------------------------------------------
__global__ __launch_bounds__(256) void out_proj(
    const unsigned short* __restrict__ ao, const unsigned short* __restrict__ wo,
    const float* __restrict__ bo, float* __restrict__ out) {
  __shared__ unsigned short tin[16][CDIM];
  const int tid = threadIdx.x;
  const int rowbase = blockIdx.x * 16;

  for (int i = tid; i < 16 * CDIM; i += 256)
    tin[i >> 7][i & 127] = ao[(size_t)rowbase * CDIM + i];
  __syncthreads();

  const int lane = tid & 31;
  const int wave = tid >> 5;          // n-tile 0..7
  const int half = lane >> 4;
  const int l15  = lane & 15;
  const int kbaseA = half * 8;
  const int kb2    = half * 16;

  v8f acc = {};
#pragma unroll
  for (int ks = 0; ks < 4; ++ks) {
    Frag a, b;
#pragma unroll
    for (int i = 0; i < 16; ++i) {
      int koff = kb2 + i;
      a.u[i] = tin[l15][ks * 32 + koff];
    }
    int ncol = wave * 16 + l15;
    const unsigned short* wrow = wo + ncol * CDIM + ks * 32 + kb2;
#pragma unroll
    for (int i = 0; i < 16; ++i) b.u[i] = wrow[i];
    acc = wmma_bf16(a.v, b.v, acc);
  }

  float bias = bo[wave * 16 + l15];
#pragma unroll
  for (int p = 0; p < 8; ++p) {
    int r = rowbase + p + half * 8;
    out[(size_t)r * CDIM + wave * 16 + l15] = acc[p] + bias;
  }
}

// ---------------------------------------------------------------------------
extern "C" void kernel_launch(void* const* d_in, const int* in_sizes, int n_in,
                              void* d_out, int out_size, void* d_ws, size_t ws_size,
                              hipStream_t stream) {
  const float* vals = (const float*)d_in[0];
  const float* keys = (const float*)d_in[1];
  const float* qry  = (const float*)d_in[2];
  const float* Wv   = (const float*)d_in[3];
  const float* bv   = (const float*)d_in[4];
  const float* Wk   = (const float*)d_in[5];
  const float* bk   = (const float*)d_in[6];
  const float* Wq   = (const float*)d_in[7];
  const float* bq   = (const float*)d_in[8];
  const float* Wo   = (const float*)d_in[9];
  const float* bo   = (const float*)d_in[10];
  float* out = (float*)d_out;

  char* ws = (char*)d_ws;
  unsigned short* wbf = (unsigned short*)ws;                 // Wv|Wk|Wq|Wo bf16
  unsigned short* vb  = (unsigned short*)(ws + 131072);      // 11,141,120 elems each
  unsigned short* kb  = vb + 11141120;
  unsigned short* qb  = kb + 11141120;
  unsigned short* ao  = qb + 11141120;

  convert_weights<<<256, 256, 0, stream>>>(Wv, Wk, Wq, Wo, wbf);
  qkv_proj<<<NROWS / 16, 256, 0, stream>>>(vals, keys, qry, wbf, bv, bk, bq,
                                           vb, kb, qb);
  attention<<<8 * 64 * HEADS, 128, 0, stream>>>(qb, kb, vb, ao);
  out_proj<<<NROWS / 16, 256, 0, stream>>>(ao, wbf + 3 * 16384, bo, out);
}
